// TwoStageAttentionLayer_18141941858339
// MI455X (gfx1250) — matrix operand order repaired and checked by reading, not compiled, not run on hardware
//
// TwoStageAttentionLayer for MI455X (gfx1250).
// All K>=256 GEMMs run through v_wmma_f32_16x16x32_bf16 (bf16 ops, f32 accum):
// compute-bound (~200 GFLOP vs ~350MB traffic at 23.3 TB/s HBM). B (weight)
// tiles are staged with GLOBAL_LOAD_ASYNC_TO_LDS_B128 (ASYNCcnt-tracked, no
// VGPR staging) overlapped with WMMA on the other LDS buffer; A (activation)
// tiles are converted f32->bf16 in flight and written with ds_store_b128.
// Tiny attention cores (DQK=3 temporal, 10-key router), softmax and LayerNorm
// stay fp32 VALU with wave32 shuffle reductions.

#include <hip/hip_runtime.h>
#include <cstdint>

#define DEVFN __device__ __forceinline__

typedef __bf16 bf16_t;
typedef __attribute__((ext_vector_type(16))) __bf16 v16bf;
typedef __attribute__((ext_vector_type(8)))  __bf16 v8bf;
typedef __attribute__((ext_vector_type(8)))  float  v8f;
typedef __attribute__((ext_vector_type(4)))  int    v4i;
typedef __attribute__((address_space(1))) v4i gv4i;   // global int4
typedef __attribute__((address_space(3))) v4i lv4i;   // LDS int4

constexpr int Bc = 4, Nn = 1362, Tt = 12, Dd = 256, Hh = 8, FACc = 10, DQKc = 3, DFFc = 1024, Ee = 32;
constexpr int MTOK = Bc * Nn * Tt;   // 65376 tokens
constexpr float EPSc = 1e-5f;

DEVFN float warp_sum(float v) {
#pragma unroll
  for (int off = 16; off > 0; off >>= 1) v += __shfl_xor(v, off, 32);
  return v;
}

DEVFN void wait_async0() { asm volatile("s_wait_asynccnt 0x0" ::: "memory"); }

// async copy 16B global -> LDS (per lane), ASYNCcnt-tracked
DEVFN void async_b128(const bf16_t* g, bf16_t* l) {
  __builtin_amdgcn_global_load_async_to_lds_b128((gv4i*)g, (lv4i*)l, 0, 0);
}

// ---------------------------------------------------------------------------
// bf16 weight conversion kernels
// ---------------------------------------------------------------------------
__global__ void cvt_bf16_k(const float* __restrict__ in, bf16_t* __restrict__ out, int n) {
  int i = blockIdx.x * 256 + threadIdx.x;
  if (i < n) out[i] = (bf16_t)in[i];
}

// W_v is (H, D, E) -> stack into (D, H*E) so syn V projection is one GEMM.
__global__ void cvt_wvstack_k(const float* __restrict__ Wv, bf16_t* __restrict__ out) {
  int i = blockIdx.x * 256 + threadIdx.x;
  if (i < Dd * Dd) {
    int f = i >> 8, c = i & 255, h = c >> 5, e = c & 31;
    out[i] = (bf16_t)Wv[(size_t)(h * Dd + f) * Ee + e];
  }
}

// ---------------------------------------------------------------------------
// WMMA bf16 GEMM:  C(M,N) = act( A(M,K) @ B(K,N) + bias )
//   A: f32, converted to bf16 while staging into LDS.
//   amap==1 gathers A rows from x-token order into stage-2 (b,t,n) row order.
//   A1 != null: concat mode, A = [A0 | A1], each with K0 columns.
//   act: 0 = none, 1 = exact GELU.
// Tile: BM=128, BN=128, BK=32; 256 threads = 8 waves; wave tile 32x64 =
// 2x4 grid of v_wmma_f32_16x16x32_bf16 accumulators.
// ---------------------------------------------------------------------------
#define BM 128
#define BN 128
#define BK 32

__global__ __launch_bounds__(256) void gemm_bf16_k(
    const float* __restrict__ A0, const float* __restrict__ A1, int K0,
    const bf16_t* __restrict__ Bw, const float* __restrict__ bias,
    float* __restrict__ C, int M, int K, int Nsz, int amap, int act)
{
  __shared__ bf16_t As[2][BM][BK + 16];   // row stride 96B (16B aligned)
  __shared__ bf16_t Bs[2][BK][BN + 16];   // row stride 288B (16B aligned)

  const int tid   = threadIdx.x;
  const int nBase = blockIdx.x * BN;
  const int mBase = blockIdx.y * BM;

  // A staging: thread -> (row 0..127, 16-col half)
  const int ra = tid >> 1;
  const int ca = (tid & 1) << 4;
  const int grow = mBase + ra;
  const bool rowok = grow < M;
  int srow = grow;
  if (amap == 1 && rowok) {
    int bt = grow / Nn, n = grow % Nn;
    int b = bt / Tt, t = bt % Tt;
    srow = (b * Nn + n) * Tt + t;   // gather from x token (b,n,t)
  }
  // B staging: thread -> (k row 0..31, 16-elem chunk); async 16B per lane
  const int rb = tid >> 3;
  const int ob = (tid & 7) << 4;

  auto load_stage = [&](int kb, int s) {
    // ---- B tile: async global -> LDS, 2 x b128 per thread
    const bf16_t* bp = Bw + (size_t)(kb * BK + rb) * Nsz + nBase + ob;
    async_b128(bp,     &Bs[s][rb][ob]);
    async_b128(bp + 8, &Bs[s][rb][ob + 8]);
    // ---- A tile (f32 -> bf16), 2 x ds_store_b128 per thread
    int kcol = kb * BK + ca;
    const float* srcA = A0;
    if (A1 != nullptr && kcol >= K0) { srcA = A1; kcol -= K0; }
    float4 f0 = {0,0,0,0}, f1 = {0,0,0,0}, f2 = {0,0,0,0}, f3 = {0,0,0,0};
    if (rowok) {
      const float4* p = (const float4*)(srcA + (size_t)srow * K0 + kcol);
      f0 = p[0]; f1 = p[1]; f2 = p[2]; f3 = p[3];
    }
    v8bf lo = { (bf16_t)f0.x, (bf16_t)f0.y, (bf16_t)f0.z, (bf16_t)f0.w,
                (bf16_t)f1.x, (bf16_t)f1.y, (bf16_t)f1.z, (bf16_t)f1.w };
    v8bf hi = { (bf16_t)f2.x, (bf16_t)f2.y, (bf16_t)f2.z, (bf16_t)f2.w,
                (bf16_t)f3.x, (bf16_t)f3.y, (bf16_t)f3.z, (bf16_t)f3.w };
    bf16_t* da = &As[s][ra][ca];
    *(v8bf*)da       = lo;
    *(v8bf*)(da + 8) = hi;
  };

  const int lane = tid & 31, wave = tid >> 5;
  const int wm = wave >> 1, wn = wave & 1;
  const int lr = lane & 15;
  const int kc = (lane < 16) ? 0 : 8;   // ISA A-frag K chunk select

  v8f acc[2][4];
#pragma unroll
  for (int mi = 0; mi < 2; ++mi)
#pragma unroll
    for (int ni = 0; ni < 4; ++ni)
#pragma unroll
      for (int i = 0; i < 8; ++i) acc[mi][ni][i] = 0.f;

  const int nK = K / BK;
  load_stage(0, 0);
  wait_async0();
  __syncthreads();

  for (int kb = 0; kb < nK; ++kb) {
    const int s = kb & 1;
    if (kb + 1 < nK) load_stage(kb + 1, s ^ 1);

    union Frag { v16bf v; v8bf h[2]; };
    Frag a0, a1, b0, b1, b2, b3;
    // A fragment: lane = row (mod 16); Ks {kc..kc+7, kc+16..kc+23}
    const bf16_t* pa0 = &As[s][wm * 32 + lr][kc];
    a0.h[0] = *(const v8bf*)pa0;  a0.h[1] = *(const v8bf*)(pa0 + 16);
    const bf16_t* pa1 = &As[s][wm * 32 + 16 + lr][kc];
    a1.h[0] = *(const v8bf*)pa1;  a1.h[1] = *(const v8bf*)(pa1 + 16);
    // B fragment: lane = N col (mod 16); Ks {kc..kc+7, kc+16..kc+23}
    const bf16_t* pb = &Bs[s][kc][wn * 64 + lr];
#pragma unroll
    for (int i = 0; i < 8; ++i) {
      b0.v[i] = pb[i * (BN + 16) +  0]; b0.v[i + 8] = pb[(i + 16) * (BN + 16) +  0];
      b1.v[i] = pb[i * (BN + 16) + 16]; b1.v[i + 8] = pb[(i + 16) * (BN + 16) + 16];
      b2.v[i] = pb[i * (BN + 16) + 32]; b2.v[i + 8] = pb[(i + 16) * (BN + 16) + 32];
      b3.v[i] = pb[i * (BN + 16) + 48]; b3.v[i + 8] = pb[(i + 16) * (BN + 16) + 48];
    }

    acc[0][0] = __builtin_amdgcn_wmma_f32_16x16x32_bf16(false, a0.v, false, b0.v, (short)0, acc[0][0], false, false);
    acc[0][1] = __builtin_amdgcn_wmma_f32_16x16x32_bf16(false, a0.v, false, b1.v, (short)0, acc[0][1], false, false);
    acc[0][2] = __builtin_amdgcn_wmma_f32_16x16x32_bf16(false, a0.v, false, b2.v, (short)0, acc[0][2], false, false);
    acc[0][3] = __builtin_amdgcn_wmma_f32_16x16x32_bf16(false, a0.v, false, b3.v, (short)0, acc[0][3], false, false);
    acc[1][0] = __builtin_amdgcn_wmma_f32_16x16x32_bf16(false, a1.v, false, b0.v, (short)0, acc[1][0], false, false);
    acc[1][1] = __builtin_amdgcn_wmma_f32_16x16x32_bf16(false, a1.v, false, b1.v, (short)0, acc[1][1], false, false);
    acc[1][2] = __builtin_amdgcn_wmma_f32_16x16x32_bf16(false, a1.v, false, b2.v, (short)0, acc[1][2], false, false);
    acc[1][3] = __builtin_amdgcn_wmma_f32_16x16x32_bf16(false, a1.v, false, b3.v, (short)0, acc[1][3], false, false);

    wait_async0();        // wave's async B loads for next stage complete
    __syncthreads();      // all waves' tiles visible
  }

  // Epilogue per ISA C/D layout: VGPR i -> row i + 8*(lane/16), col = lane%16
#pragma unroll
  for (int mi = 0; mi < 2; ++mi)
#pragma unroll
    for (int ni = 0; ni < 4; ++ni) {
      const int col = nBase + wn * 64 + ni * 16 + (lane & 15);
      const float bv = bias ? bias[col] : 0.f;
      const int rbase = mBase + wm * 32 + mi * 16 + ((lane >> 4) << 3);
#pragma unroll
      for (int i = 0; i < 8; ++i) {
        const int row = rbase + i;
        if (row < M) {
          float v = acc[mi][ni][i] + bv;
          if (act == 1) v = 0.5f * v * (1.f + erff(v * 0.70710678118f));
          C[(size_t)row * Nsz + col] = v;
        }
      }
    }
}

// ---------------------------------------------------------------------------
// Stage-1 synaptic temporal attention (per node): q = (x@Wu)*(x@Ws[n]),
// k = (x@Wk) reinterpreted (T,DQK)->(DQK,T), att = softmax(qk/sqrt(3)),
// out = att @ v, + residual, LN1.
// ---------------------------------------------------------------------------
__global__ __launch_bounds__(256) void syn_attn_k(
    const float* __restrict__ x,  const float* __restrict__ Wu,
    const float* __restrict__ Ws, const float* __restrict__ Wk,
    const float* __restrict__ Vsyn, const float* __restrict__ g1,
    const float* __restrict__ be1, float* __restrict__ y1)
{
  __shared__ float xs[Tt][Dd];
  __shared__ float vs[Tt][Dd];
  __shared__ float qv[Hh][Tt][DQKc];
  __shared__ float kv[Hh][Tt][DQKc];
  __shared__ float att[Hh][Tt][Tt];
  __shared__ float ob[Tt][Dd];

  const int node = blockIdx.x;
  const int n = node % Nn;
  const int tid = threadIdx.x;
  const size_t base = (size_t)node * Tt * Dd;

  for (int i = tid; i < Tt * Dd; i += 256) {
    xs[i >> 8][i & 255] = x[base + i];
    vs[i >> 8][i & 255] = Vsyn[base + i];
  }
  __syncthreads();

  // skinny projections: per (h,t,dq) dot over D=256
  for (int o = tid; o < Hh * Tt * DQKc; o += 256) {
    const int h = o / (Tt * DQKc);
    const int r = o % (Tt * DQKc);
    const int t = r / DQKc, dq = r % DQKc;
    float su = 0.f, ss = 0.f, sk = 0.f;
    for (int f = 0; f < Dd; ++f) {
      const float xv = xs[t][f];
      const int wi = h * Dd + f;
      su += xv * Wu[wi * DQKc + dq];
      ss += xv * Ws[((size_t)wi * Nn + n) * DQKc + dq];
      sk += xv * Wk[wi * DQKc + dq];
    }
    qv[h][t][dq] = su * ss;
    kv[h][t][dq] = sk;
  }
  __syncthreads();

  // scores with faithful (T,DQK)->(DQK,T) reinterpretation of k
  const float qscale = 0.57735026919f;  // 1/sqrt(DQK)
  for (int o = tid; o < Hh * Tt * Tt; o += 256) {
    const int h = o / (Tt * Tt);
    const int r = o % (Tt * Tt);
    const int ti = r / Tt, j = r % Tt;
    float s = 0.f;
#pragma unroll
    for (int d = 0; d < DQKc; ++d) {
      const int m = d * Tt + j;
      s += qv[h][ti][d] * kv[h][m / DQKc][m % DQKc];
    }
    att[h][ti][j] = s * qscale;
  }
  __syncthreads();

  for (int o = tid; o < Hh * Tt; o += 256) {
    const int h = o / Tt, ti = o % Tt;
    float mx = -1e30f;
    for (int j = 0; j < Tt; ++j) mx = fmaxf(mx, att[h][ti][j]);
    float sm = 0.f;
    for (int j = 0; j < Tt; ++j) { float e = __expf(att[h][ti][j] - mx); att[h][ti][j] = e; sm += e; }
    const float inv = 1.f / sm;
    for (int j = 0; j < Tt; ++j) att[h][ti][j] *= inv;
  }
  __syncthreads();

  for (int o = tid; o < Tt * Dd; o += 256) {
    const int t = o >> 8, c = o & 255, h = c >> 5;
    float s = 0.f;
#pragma unroll
    for (int j = 0; j < Tt; ++j) s += att[h][t][j] * vs[j][c];
    ob[t][c] = s + xs[t][c];   // + residual st_in
  }
  __syncthreads();

  // LN1: one wave per row
  const int lane = tid & 31, wave = tid >> 5;
  for (int t = wave; t < Tt; t += 8) {
    float vals[8], s = 0.f, s2 = 0.f;
#pragma unroll
    for (int j = 0; j < 8; ++j) {
      const float v = ob[t][lane + 32 * j];
      vals[j] = v; s += v; s2 += v * v;
    }
    s = warp_sum(s); s2 = warp_sum(s2);
    const float mean = s * (1.f / Dd);
    const float inv = rsqrtf(s2 * (1.f / Dd) - mean * mean + EPSc);
#pragma unroll
    for (int j = 0; j < 8; ++j) {
      const int c = lane + 32 * j;
      y1[base + (size_t)t * Dd + c] = (vals[j] - mean) * inv * g1[c] + be1[c];
    }
  }
}

// ---------------------------------------------------------------------------
// Residual + LayerNorm over D=256; one wave per row; bmap==1 gathers the
// residual from x in stage-2 row order.
// ---------------------------------------------------------------------------
__global__ __launch_bounds__(256) void ln_res_k(
    const float* __restrict__ a, const float* __restrict__ bsrc, int bmap,
    const float* __restrict__ g, const float* __restrict__ be,
    float* __restrict__ out, int M)
{
  const int r = blockIdx.x * 8 + (threadIdx.x >> 5);
  if (r >= M) return;
  const int lane = threadIdx.x & 31;
  size_t br = (size_t)r;
  if (bmap == 1) {
    const int bt = r / Nn, n = r % Nn;
    const int b = bt / Tt, t = bt % Tt;
    br = (size_t)((b * Nn + n) * Tt + t);
  }
  float vals[8], s = 0.f, s2 = 0.f;
#pragma unroll
  for (int j = 0; j < 8; ++j) {
    const int c = lane + 32 * j;
    const float v = a[(size_t)r * Dd + c] + bsrc[br * Dd + c];
    vals[j] = v; s += v; s2 += v * v;
  }
  s = warp_sum(s); s2 = warp_sum(s2);
  const float mean = s * (1.f / Dd);
  const float inv = rsqrtf(s2 * (1.f / Dd) - mean * mean + EPSc);
#pragma unroll
  for (int j = 0; j < 8; ++j) {
    const int c = lane + 32 * j;
    out[(size_t)r * Dd + c] = (vals[j] - mean) * inv * g[c] + be[c];
  }
}

// ---------------------------------------------------------------------------
// Small fp32 GEMM (K=N=256), one row per block. Used for the 120/480-row
// router projections (deck Q, deck Wo, rcv K/V).
// ---------------------------------------------------------------------------
__global__ __launch_bounds__(256) void sgemm256_k(
    const float* __restrict__ A, const float* __restrict__ W,
    const float* __restrict__ bias, float* __restrict__ C, int M)
{
  __shared__ float ar[Dd];
  const int row = blockIdx.x;
  const int tid = threadIdx.x;
  if (row >= M) return;
  ar[tid] = A[(size_t)row * Dd + tid];
  __syncthreads();
  float acc = bias ? bias[tid] : 0.f;
#pragma unroll 4
  for (int k = 0; k < Dd; ++k) acc += ar[k] * W[(size_t)k * Dd + tid];
  C[(size_t)row * Dd + tid] = acc;
}

// ---------------------------------------------------------------------------
// snd router attention: per (b,t) block, L=10 queries, stream S=1362 keys with
// online softmax. wave = head (wave32 lanes = head dim 32).
// ---------------------------------------------------------------------------
__global__ __launch_bounds__(256) void snd_attn_k(
    const float* __restrict__ Qs, const float* __restrict__ Kf,
    const float* __restrict__ Vf, float* __restrict__ aout)
{
  const int bt = blockIdx.x;
  const int t = bt % Tt;
  const int lane = threadIdx.x & 31, h = threadIdx.x >> 5;
  const int col = h * Ee + lane;

  float q[FACc], m[FACc], d[FACc], acc[FACc];
#pragma unroll
  for (int l = 0; l < FACc; ++l) {
    q[l] = Qs[(size_t)(t * FACc + l) * Dd + col];
    m[l] = -1e30f; d[l] = 0.f; acc[l] = 0.f;
  }
  const float scl = 0.17677669529f;  // 1/sqrt(E)
  const size_t rbase = (size_t)bt * Nn;
  for (int s = 0; s < Nn; ++s) {
    __builtin_prefetch(Kf + (rbase + s + 16) * Dd + col, 0, 1);
    __builtin_prefetch(Vf + (rbase + s + 16) * Dd + col, 0, 1);
    const float kvv = Kf[(rbase + s) * Dd + col];
    const float vvv = Vf[(rbase + s) * Dd + col];
#pragma unroll
    for (int l = 0; l < FACc; ++l) {
      const float p = warp_sum(q[l] * kvv) * scl;
      const float mn = fmaxf(m[l], p);
      const float corr = __expf(m[l] - mn);
      const float e = __expf(p - mn);
      d[l] = d[l] * corr + e;
      acc[l] = acc[l] * corr + e * vvv;
      m[l] = mn;
    }
  }
#pragma unroll
  for (int l = 0; l < FACc; ++l)
    aout[(size_t)(bt * FACc + l) * Dd + col] = acc[l] / d[l];
}

// ---------------------------------------------------------------------------
// rcv attention: per stage-2 row, 10 deck keys. wave = head.
// ---------------------------------------------------------------------------
__global__ __launch_bounds__(256) void rcv_attn_k(
    const float* __restrict__ Qr, const float* __restrict__ Kd,
    const float* __restrict__ Vd, float* __restrict__ aout)
{
  const int r = blockIdx.x;
  const int bt = r / Nn;
  const int lane = threadIdx.x & 31, h = threadIdx.x >> 5;
  const int col = h * Ee + lane;
  const float q = Qr[(size_t)r * Dd + col];
  const float scl = 0.17677669529f;
  float sc[FACc];
  float mx = -1e30f;
#pragma unroll
  for (int s = 0; s < FACc; ++s) {
    const float kvv = Kd[(size_t)(bt * FACc + s) * Dd + col];
    sc[s] = warp_sum(q * kvv) * scl;
    mx = fmaxf(mx, sc[s]);
  }
  float sum = 0.f, o = 0.f;
#pragma unroll
  for (int s = 0; s < FACc; ++s) {
    const float e = __expf(sc[s] - mx);
    sum += e;
    o += e * Vd[(size_t)(bt * FACc + s) * Dd + col];
  }
  aout[(size_t)r * Dd + col] = o / sum;
}

// ---------------------------------------------------------------------------
extern "C" void kernel_launch(void* const* d_in, const int* in_sizes, int n_in,
                              void* d_out, int out_size, void* d_ws, size_t ws_size,
                              hipStream_t stream)
{
  (void)in_sizes; (void)n_in; (void)out_size; (void)ws_size;
  const float* x       = (const float*)d_in[0];
  const float* W_u     = (const float*)d_in[1];
  const float* W_s     = (const float*)d_in[2];
  const float* W_ksyn  = (const float*)d_in[3];
  const float* W_v     = (const float*)d_in[4];
  const float* deck    = (const float*)d_in[5];
  const float* proj_W  = (const float*)d_in[6];
  const float* proj_b  = (const float*)d_in[7];
  const float* snd_Wq  = (const float*)d_in[8];
  const float* snd_Wk  = (const float*)d_in[9];
  const float* snd_Wv  = (const float*)d_in[10];
  const float* snd_Wo  = (const float*)d_in[11];
  const float* snd_bq  = (const float*)d_in[12];
  const float* snd_bk  = (const float*)d_in[13];
  const float* snd_bv  = (const float*)d_in[14];
  const float* snd_bo  = (const float*)d_in[15];
  const float* rcv_Wq  = (const float*)d_in[16];
  const float* rcv_Wk  = (const float*)d_in[17];
  const float* rcv_Wv  = (const float*)d_in[18];
  const float* rcv_Wo  = (const float*)d_in[19];
  const float* rcv_bq  = (const float*)d_in[20];
  const float* rcv_bk  = (const float*)d_in[21];
  const float* rcv_bv  = (const float*)d_in[22];
  const float* rcv_bo  = (const float*)d_in[23];
  const float* m1_W1   = (const float*)d_in[24];
  const float* m1_b1   = (const float*)d_in[25];
  const float* m1_W2   = (const float*)d_in[26];
  const float* m1_b2   = (const float*)d_in[27];
  const float* m2_W1   = (const float*)d_in[28];
  const float* m2_b1   = (const float*)d_in[29];
  const float* m2_W2   = (const float*)d_in[30];
  const float* m2_b2   = (const float*)d_in[31];
  const float* ln1_g   = (const float*)d_in[32];
  const float* ln1_b   = (const float*)d_in[33];
  const float* ln2_g   = (const float*)d_in[34];
  const float* ln2_b   = (const float*)d_in[35];
  const float* ln3_g   = (const float*)d_in[36];
  const float* ln3_b   = (const float*)d_in[37];
  const float* ln4_g   = (const float*)d_in[38];
  const float* ln4_b   = (const float*)d_in[39];

  // ---- workspace carve
  size_t off = 0;
  auto carve = [&](size_t bytes) -> void* {
    off = (off + 255) & ~(size_t)255;
    void* p = (char*)d_ws + off;
    off += bytes;
    return p;
  };
  bf16_t* bWv    = (bf16_t*)carve((size_t)Dd * Dd * 2);
  bf16_t* bM1W1  = (bf16_t*)carve((size_t)Dd * DFFc * 2);
  bf16_t* bM1W2  = (bf16_t*)carve((size_t)DFFc * Dd * 2);
  bf16_t* bM2W1  = (bf16_t*)carve((size_t)Dd * DFFc * 2);
  bf16_t* bM2W2  = (bf16_t*)carve((size_t)DFFc * Dd * 2);
  bf16_t* bSndWk = (bf16_t*)carve((size_t)Dd * Dd * 2);
  bf16_t* bSndWv = (bf16_t*)carve((size_t)Dd * Dd * 2);
  bf16_t* bRcvWq = (bf16_t*)carve((size_t)Dd * Dd * 2);
  bf16_t* bRcvWo = (bf16_t*)carve((size_t)Dd * Dd * 2);
  bf16_t* bProj  = (bf16_t*)carve((size_t)2 * Dd * Dd * 2);
  float* bufA = (float*)carve((size_t)MTOK * Dd * 4);
  float* bufB = (float*)carve((size_t)MTOK * Dd * 4);
  float* bufC = (float*)carve((size_t)MTOK * Dd * 4);
  float* bufH = (float*)carve((size_t)MTOK * DFFc * 4);
  float* qsnd = (float*)carve((size_t)Tt * FACc * Dd * 4);
  float* attd = (float*)carve((size_t)Bc * Tt * FACc * Dd * 4);
  float* dbuf = (float*)carve((size_t)Bc * Tt * FACc * Dd * 4);
  float* kd   = (float*)carve((size_t)Bc * Tt * FACc * Dd * 4);
  float* vd   = (float*)carve((size_t)Bc * Tt * FACc * Dd * 4);

  const dim3 blk(256);
  const int gy = (MTOK + BM - 1) / BM;

  auto cvt = [&](const float* src, bf16_t* dst, int n) {
    cvt_bf16_k<<<(n + 255) / 256, blk, 0, stream>>>(src, dst, n);
  };
  cvt_wvstack_k<<<(Dd * Dd + 255) / 256, blk, 0, stream>>>(W_v, bWv);
  cvt(m1_W1, bM1W1, Dd * DFFc);
  cvt(m1_W2, bM1W2, DFFc * Dd);
  cvt(m2_W1, bM2W1, Dd * DFFc);
  cvt(m2_W2, bM2W2, DFFc * Dd);
  cvt(snd_Wk, bSndWk, Dd * Dd);
  cvt(snd_Wv, bSndWv, Dd * Dd);
  cvt(rcv_Wq, bRcvWq, Dd * Dd);
  cvt(rcv_Wo, bRcvWo, Dd * Dd);
  cvt(proj_W, bProj, 2 * Dd * Dd);

  // ---- stage 1
  // vsyn = x @ WvStack
  gemm_bf16_k<<<dim3(Dd / BN, gy), blk, 0, stream>>>(x, nullptr, Dd, bWv, nullptr, bufA, MTOK, Dd, Dd, 0, 0);
  // per-node syn attention + residual + LN1 -> y1
  syn_attn_k<<<Bc * Nn, blk, 0, stream>>>(x, W_u, W_s, W_ksyn, bufA, ln1_g, ln1_b, bufB);
  // MLP1
  gemm_bf16_k<<<dim3(DFFc / BN, gy), blk, 0, stream>>>(bufB, nullptr, Dd, bM1W1, m1_b1, bufH, MTOK, Dd, DFFc, 0, 1);
  gemm_bf16_k<<<dim3(Dd / BN, gy), blk, 0, stream>>>(bufH, nullptr, DFFc, bM1W2, m1_b2, bufC, MTOK, DFFc, Dd, 0, 0);
  // time_enc = LN2(mlp_out + y1) -> bufC
  ln_res_k<<<(MTOK + 7) / 8, blk, 0, stream>>>(bufC, bufB, 0, ln2_g, ln2_b, bufC, MTOK);

  // ---- stage 2 (all row-major in stage-2 order r = (b*T+t)*N + n)
  gemm_bf16_k<<<dim3(Dd / BN, gy), blk, 0, stream>>>(x, nullptr, Dd, bSndWk, snd_bk, bufA, MTOK, Dd, Dd, 1, 0);
  gemm_bf16_k<<<dim3(Dd / BN, gy), blk, 0, stream>>>(x, nullptr, Dd, bSndWv, snd_bv, bufB, MTOK, Dd, Dd, 1, 0);
  sgemm256_k<<<Tt * FACc, blk, 0, stream>>>(deck, snd_Wq, snd_bq, qsnd, Tt * FACc);
  snd_attn_k<<<Bc * Tt, blk, 0, stream>>>(qsnd, bufA, bufB, attd);
  sgemm256_k<<<Bc * Tt * FACc, blk, 0, stream>>>(attd, snd_Wo, snd_bo, dbuf, Bc * Tt * FACc);
  sgemm256_k<<<Bc * Tt * FACc, blk, 0, stream>>>(dbuf, rcv_Wk, rcv_bk, kd, Bc * Tt * FACc);
  sgemm256_k<<<Bc * Tt * FACc, blk, 0, stream>>>(dbuf, rcv_Wv, rcv_bv, vd, Bc * Tt * FACc);
  gemm_bf16_k<<<dim3(Dd / BN, gy), blk, 0, stream>>>(x, nullptr, Dd, bRcvWq, rcv_bq, bufA, MTOK, Dd, Dd, 1, 0);
  rcv_attn_k<<<MTOK, blk, 0, stream>>>(bufA, kd, vd, bufB);
  gemm_bf16_k<<<dim3(Dd / BN, gy), blk, 0, stream>>>(bufB, nullptr, Dd, bRcvWo, rcv_bo, bufA, MTOK, Dd, Dd, 0, 0);
  // z = LN3(rcvO + spa_in)
  ln_res_k<<<(MTOK + 7) / 8, blk, 0, stream>>>(bufA, x, 1, ln3_g, ln3_b, bufA, MTOK);
  // MLP2
  gemm_bf16_k<<<dim3(DFFc / BN, gy), blk, 0, stream>>>(bufA, nullptr, Dd, bM2W1, m2_b1, bufH, MTOK, Dd, DFFc, 0, 1);
  gemm_bf16_k<<<dim3(Dd / BN, gy), blk, 0, stream>>>(bufH, nullptr, DFFc, bM2W2, m2_b2, bufB, MTOK, DFFc, Dd, 0, 0);
  // spa_enc = LN4(mlp_out + z) -> bufB
  ln_res_k<<<(MTOK + 7) / 8, blk, 0, stream>>>(bufB, bufA, 0, ln4_g, ln4_b, bufB, MTOK);

  // ---- final: out = concat(time_enc, spa_enc) @ proj_W + proj_b
  // (stage-2 row order coincides with output (b,n,t) row order, so plain GEMM)
  gemm_bf16_k<<<dim3(Dd / BN, gy), blk, 0, stream>>>(bufC, bufB, Dd, bProj, proj_b, (float*)d_out, MTOK, 2 * Dd, Dd, 0, 0);
}
